// Attention_30889404792851
// MI455X (gfx1250) — compile-verified, matrix operand order checked by reading, not run on hardware
//
#include <hip/hip_runtime.h>
#include <hip/hip_bf16.h>

// ---------------------------------------------------------------------------
// MHA forward for MI455X (gfx1250, wave32, WMMA 16x16x32 f16 path)
//   x:[2048,4,1024] f32, w_qkv:[3072,1024], w_out:[1024,1024], ln_w/ln_b:[1024]
//   out = x + Proj(Softmax(Q K^T / sqrt(1024)) V)
// GEMMs + attention run on v_wmma_f32_16x16x32_f16; shared tiles are staged
// into LDS with the CDNA5 async global->LDS copies (ASYNCcnt) and double
// buffering so copies overlap the WMMA stream.
// ---------------------------------------------------------------------------

typedef __attribute__((ext_vector_type(16))) _Float16 v16h;
typedef __attribute__((ext_vector_type(8)))  _Float16 h8;
typedef __attribute__((ext_vector_type(8)))  float    v8f;

#define SEQ     2048
#define BSZ     4
#define TOK     (SEQ*BSZ)      // 8192
#define RES     1024
#define NH      16
#define HD      64
#define QKVD    (NH*3*HD)      // 3072
#define LN_EPS  1e-5f
#define SM_SCALE 0.03125f      // 1/sqrt(1024)

__device__ __forceinline__ v8f wmma_f16(v16h a, v16h b, v8f c) {
    return __builtin_amdgcn_wmma_f32_16x16x32_f16(
        /*neg_a=*/false, a, /*neg_b=*/false, b,
        /*c_mod=*/(short)0, c, /*reuse_a=*/false, /*reuse_b=*/false);
}

// 16-bit A/B^T fragment for 16x16x32: p points at (row, kbase) for this lane,
// kbase already includes the (lane>>4)*8 shift. Elems [0..7]=K kbase..+7,
// elems [8..15]=K kbase+16..+23.
__device__ __forceinline__ v16h frag16(const _Float16* __restrict__ p) {
    h8 lo = *(const h8*)p;
    h8 hi = *(const h8*)(p + 16);
    v16h f;
#pragma unroll
    for (int i = 0; i < 8; ++i) { f[i] = lo[i]; f[i + 8] = hi[i]; }
    return f;
}

// CDNA5 async copy: 16 bytes global -> LDS, tracked by ASYNCcnt.
// LDS address = low 32 bits of the flat shared pointer (LDS aperture).
__device__ __forceinline__ void async_ld16(const _Float16* g, _Float16* l) {
    unsigned lds = (unsigned)(uintptr_t)l;
    unsigned long long ga = (unsigned long long)(uintptr_t)g;
    asm volatile("global_load_async_to_lds_b128 %0, %1, off"
                 :: "v"(lds), "v"(ga) : "memory");
}
__device__ __forceinline__ void wait_async0() {
    asm volatile("s_wait_asynccnt 0x0" ::: "memory");
}

// ---------------------------------------------------------------------------
// 1) LayerNorm -> f16
// ---------------------------------------------------------------------------
__global__ void __launch_bounds__(256)
ln_f16_kernel(const float* __restrict__ x, const float* __restrict__ w,
              const float* __restrict__ b, _Float16* __restrict__ y) {
    __shared__ float red[16];
    const int t = blockIdx.x;
    const float* row = x + (size_t)t * RES;
    float s = 0.f, s2 = 0.f;
#pragma unroll
    for (int i = 0; i < RES / 256; ++i) {
        float v = row[threadIdx.x + i * 256];
        s += v; s2 += v * v;
    }
#pragma unroll
    for (int m = 16; m >= 1; m >>= 1) { s += __shfl_xor(s, m); s2 += __shfl_xor(s2, m); }
    const int wv = threadIdx.x >> 5, lane = threadIdx.x & 31;
    if (lane == 0) { red[wv] = s; red[8 + wv] = s2; }
    __syncthreads();
    if (threadIdx.x == 0) {
        float a = 0.f, q = 0.f;
#pragma unroll
        for (int i = 0; i < 8; ++i) { a += red[i]; q += red[8 + i]; }
        float mu = a * (1.f / RES);
        float var = q * (1.f / RES) - mu * mu;
        red[0] = mu; red[1] = rsqrtf(var + LN_EPS);
    }
    __syncthreads();
    const float mu = red[0], inv = red[1];
#pragma unroll
    for (int i = 0; i < RES / 256; ++i) {
        int c = threadIdx.x + i * 256;
        y[(size_t)t * RES + c] = (_Float16)((row[c] - mu) * inv * w[c] + b[c]);
    }
}

// ---------------------------------------------------------------------------
// 2) f32 -> f16 (weights)
// ---------------------------------------------------------------------------
__global__ void __launch_bounds__(256)
cvt_f16_kernel(const float* __restrict__ src, _Float16* __restrict__ dst, int n) {
    int i = blockIdx.x * 256 + threadIdx.x;
    if (i < n) dst[i] = (_Float16)src[i];
}

// ---------------------------------------------------------------------------
// 3) QKV GEMM: C[TOK,QKVD](f16) = A[TOK,RES] @ B[QKVD,RES]^T
//    128 thr = 4 waves, block tile 64x64, wave tile 16x64.
//    B tile (64 rows x 32 halves = 4KB) staged via async->LDS, double buffer.
// ---------------------------------------------------------------------------
__global__ void __launch_bounds__(128)
qkv_gemm_kernel(const _Float16* __restrict__ A, const _Float16* __restrict__ B,
                _Float16* __restrict__ C) {
    __shared__ __align__(16) _Float16 bsh[2][64][32];
    const int tid = threadIdx.x;
    const int lane = tid & 31, wv = tid >> 5;
    const int mBase = blockIdx.x * 64 + wv * 16;
    const int nBase = blockIdx.y * 64;
    const int r16 = lane & 15, kb = (lane >> 4) << 3;

    // 256 16B chunks per tile -> 2 per thread: chunk q -> row q>>2, col (q&3)*8
    const int r0 = tid >> 2,          c0 = (tid & 3) << 3;
    const int r1 = (tid + 128) >> 2,  c1 = ((tid + 128) & 3) << 3;
    auto stage = [&](int k0, int buf) {
        async_ld16(B + (size_t)(nBase + r0) * RES + k0 + c0, &bsh[buf][r0][c0]);
        async_ld16(B + (size_t)(nBase + r1) * RES + k0 + c1, &bsh[buf][r1][c1]);
    };

    stage(0, 0);
    wait_async0();
    __syncthreads();

    v8f acc[4] = {v8f{}, v8f{}, v8f{}, v8f{}};
    const _Float16* ap = A + (size_t)(mBase + r16) * RES + kb;
    const int NK = RES / 32;
    for (int i = 0; i < NK; ++i) {
        const int buf = i & 1;
        if (i + 1 < NK) stage((i + 1) * 32, buf ^ 1);
        v16h af = frag16(ap + i * 32);
#pragma unroll
        for (int t = 0; t < 4; ++t)
            acc[t] = wmma_f16(af, frag16(&bsh[buf][t * 16 + r16][kb]), acc[t]);
        wait_async0();
        __syncthreads();
    }
    const int rowB = mBase + ((lane >> 4) << 3);
#pragma unroll
    for (int t = 0; t < 4; ++t)
#pragma unroll
        for (int e = 0; e < 8; ++e)
            C[(size_t)(rowB + e) * QKVD + nBase + t * 16 + r16] = (_Float16)acc[t][e];
}

// ---------------------------------------------------------------------------
// 4) V transpose: vt[(b*NH+h)][d][j] = qkv[token(j,b)][h*192+128+d]
// ---------------------------------------------------------------------------
__global__ void __launch_bounds__(256)
vtrans_kernel(const _Float16* __restrict__ qkv, _Float16* __restrict__ vt) {
    unsigned id = blockIdx.x * 256u + threadIdx.x;
    unsigned j = id & (SEQ - 1); id >>= 11;
    unsigned d = id & (HD - 1);  id >>= 6;
    unsigned h = id & (NH - 1);  unsigned b = id >> 4;
    size_t src = (size_t)(j * BSZ + b) * QKVD + h * (3 * HD) + 2 * HD + d;
    size_t dst = (((size_t)(b * NH + h)) * HD + d) * SEQ + j;
    vt[dst] = qkv[src];
}

// ---------------------------------------------------------------------------
// 5) Flash attention: wave = 16 query rows, block = 8 waves = 128 rows.
//    grid = (SEQ/128, NH, BSZ). K/V tiles (4KB each) shared by all 8 waves:
//    staged with async->LDS double buffering; P transposed C->A layout via
//    per-wave LDS slice (intra-wave DS ordering, no barrier needed).
// ---------------------------------------------------------------------------
__global__ void __launch_bounds__(256)
flash_kernel(const _Float16* __restrict__ qkv, const _Float16* __restrict__ vt,
             _Float16* __restrict__ attn) {
    __shared__ __align__(16) _Float16 ksh[2][32][64];  // [key 32][d 64]
    __shared__ __align__(16) _Float16 vsh[2][64][32];  // [d 64][key 32]
    __shared__ __align__(16) _Float16 plds[8][16][32]; // per-wave P tile
    const int tid = threadIdx.x;
    const int lane = tid & 31, wv = tid >> 5;
    const int b = blockIdx.z, h = blockIdx.y;
    const int i0 = blockIdx.x * 128 + wv * 16;
    const int r16 = lane & 15, hi = lane >> 4, kb = hi << 3;

    const _Float16* kbp = qkv + h * (3 * HD) + HD;
    const _Float16* vtb = vt + ((size_t)(b * NH + h)) * HD * SEQ;

    // 256 threads, one 16B K-chunk + one 16B V-chunk each per tile
    const int kjl = tid >> 3, kc = (tid & 7) << 3;   // ksh row 0..31, col 0..56
    const int vdr = tid >> 2, vc = (tid & 3) << 3;   // vsh row 0..63, col 0..24
    auto stage = [&](int j0, int buf) {
        async_ld16(kbp + (size_t)((j0 + kjl) * BSZ + b) * QKVD + kc,
                   &ksh[buf][kjl][kc]);
        async_ld16(vtb + (size_t)vdr * SEQ + j0 + vc, &vsh[buf][vdr][vc]);
    };

    // Q fragments (constant over key loop)
    const _Float16* qp =
        qkv + (size_t)((i0 + r16) * BSZ + b) * QKVD + h * (3 * HD) + kb;
    const v16h qa0 = frag16(qp);        // d 0..31
    const v16h qa1 = frag16(qp + 32);   // d 32..63

    float m[8], l[8];
    v8f o[4] = {v8f{}, v8f{}, v8f{}, v8f{}};
#pragma unroll
    for (int e = 0; e < 8; ++e) { m[e] = -1e30f; l[e] = 0.f; }

    stage(0, 0);
    wait_async0();
    __syncthreads();

    const int NT = SEQ / 32;
    for (int it = 0; it < NT; ++it) {
        const int buf = it & 1;
        if (it + 1 < NT) stage((it + 1) * 32, buf ^ 1);

        // ---- scores: two 16x16 tiles (K reduced over d=64, 2 WMMAs each)
        v8f s[2];
#pragma unroll
        for (int u = 0; u < 2; ++u) {
            const _Float16* kp = &ksh[buf][u * 16 + r16][kb];
            v8f c = v8f{};
            c = wmma_f16(qa0, frag16(kp), c);
            c = wmma_f16(qa1, frag16(kp + 32), c);
            s[u] = c;
        }
        // ---- online softmax (row = e + 8*hi, replicated across 16 lanes)
        float p0[8], p1[8];
#pragma unroll
        for (int e = 0; e < 8; ++e) {
            float v0 = s[0][e] * SM_SCALE, v1 = s[1][e] * SM_SCALE;
            float mx = fmaxf(v0, v1);
#pragma unroll
            for (int msk = 1; msk <= 8; msk <<= 1) mx = fmaxf(mx, __shfl_xor(mx, msk));
            float mn = fmaxf(m[e], mx);
            float corr = __expf(m[e] - mn);
            float e0 = __expf(v0 - mn), e1 = __expf(v1 - mn);
            float ps = e0 + e1;
#pragma unroll
            for (int msk = 1; msk <= 8; msk <<= 1) ps += __shfl_xor(ps, msk);
            l[e] = l[e] * corr + ps;
            m[e] = mn;
#pragma unroll
            for (int t = 0; t < 4; ++t) o[t][e] *= corr;
            p0[e] = e0; p1[e] = e1;
        }
        // ---- P: C-layout -> per-wave LDS slice -> A-layout (intra-wave only,
        //      DS ops are in-order within a wave, so no barrier required)
#pragma unroll
        for (int e = 0; e < 8; ++e) {
            plds[wv][e + 8 * hi][r16]      = (_Float16)p0[e];
            plds[wv][e + 8 * hi][16 + r16] = (_Float16)p1[e];
        }
        v16h pa = frag16(&plds[wv][r16][kb]);
        // ---- O += P @ V from staged V tile
#pragma unroll
        for (int t = 0; t < 4; ++t)
            o[t] = wmma_f16(pa, frag16(&vsh[buf][t * 16 + r16][kb]), o[t]);

        wait_async0();      // prefetch for next tile landed
        __syncthreads();    // everyone done reading buf before it refills
    }
    // ---- normalize + store f16 attn_out[token][h*64+d]
#pragma unroll
    for (int e = 0; e < 8; ++e) {
        const float inv = 1.f / l[e];
        const int tok = (i0 + e + 8 * hi) * BSZ + b;
#pragma unroll
        for (int t = 0; t < 4; ++t)
            attn[(size_t)tok * RES + h * HD + t * 16 + r16] = (_Float16)(o[t][e] * inv);
    }
}

// ---------------------------------------------------------------------------
// 6) Out projection + residual (same async-staged GEMM, f32 store)
// ---------------------------------------------------------------------------
__global__ void __launch_bounds__(128)
outproj_kernel(const _Float16* __restrict__ A, const _Float16* __restrict__ B,
               const float* __restrict__ x, float* __restrict__ out) {
    __shared__ __align__(16) _Float16 bsh[2][64][32];
    const int tid = threadIdx.x;
    const int lane = tid & 31, wv = tid >> 5;
    const int mBase = blockIdx.x * 64 + wv * 16;
    const int nBase = blockIdx.y * 64;
    const int r16 = lane & 15, kb = (lane >> 4) << 3;

    const int r0 = tid >> 2,          c0 = (tid & 3) << 3;
    const int r1 = (tid + 128) >> 2,  c1 = ((tid + 128) & 3) << 3;
    auto stage = [&](int k0, int buf) {
        async_ld16(B + (size_t)(nBase + r0) * RES + k0 + c0, &bsh[buf][r0][c0]);
        async_ld16(B + (size_t)(nBase + r1) * RES + k0 + c1, &bsh[buf][r1][c1]);
    };

    stage(0, 0);
    wait_async0();
    __syncthreads();

    v8f acc[4] = {v8f{}, v8f{}, v8f{}, v8f{}};
    const _Float16* ap = A + (size_t)(mBase + r16) * RES + kb;
    const int NK = RES / 32;
    for (int i = 0; i < NK; ++i) {
        const int buf = i & 1;
        if (i + 1 < NK) stage((i + 1) * 32, buf ^ 1);
        v16h af = frag16(ap + i * 32);
#pragma unroll
        for (int t = 0; t < 4; ++t)
            acc[t] = wmma_f16(af, frag16(&bsh[buf][t * 16 + r16][kb]), acc[t]);
        wait_async0();
        __syncthreads();
    }
    const int rowB = mBase + ((lane >> 4) << 3);
#pragma unroll
    for (int t = 0; t < 4; ++t)
#pragma unroll
        for (int e = 0; e < 8; ++e) {
            size_t idx = (size_t)(rowB + e) * RES + nBase + t * 16 + r16;
            out[idx] = x[idx] + acc[t][e];
        }
}

// ---------------------------------------------------------------------------
extern "C" void kernel_launch(void* const* d_in, const int* in_sizes, int n_in,
                              void* d_out, int out_size, void* d_ws, size_t ws_size,
                              hipStream_t stream) {
    const float* x     = (const float*)d_in[0];
    const float* w_qkv = (const float*)d_in[1];
    const float* w_out = (const float*)d_in[2];
    const float* ln_w  = (const float*)d_in[3];
    const float* ln_b  = (const float*)d_in[4];
    float* out = (float*)d_out;

    char* ws = (char*)d_ws;
    _Float16* xlnH  = (_Float16*)ws; ws += (size_t)TOK * RES * 2;           // 16 MB
    _Float16* wqkvH = (_Float16*)ws; ws += (size_t)QKVD * RES * 2;          //  6 MB
    _Float16* woutH = (_Float16*)ws; ws += (size_t)RES * RES * 2;           //  2 MB
    _Float16* qkvH  = (_Float16*)ws; ws += (size_t)TOK * QKVD * 2;          // 48 MB
    _Float16* vtH   = (_Float16*)ws; ws += (size_t)BSZ * NH * HD * SEQ * 2; // 16 MB
    _Float16* attnH = (_Float16*)ws;                                        // 16 MB

    ln_f16_kernel<<<TOK, 256, 0, stream>>>(x, ln_w, ln_b, xlnH);
    cvt_f16_kernel<<<(QKVD * RES) / 256, 256, 0, stream>>>(w_qkv, wqkvH, QKVD * RES);
    cvt_f16_kernel<<<(RES * RES) / 256, 256, 0, stream>>>(w_out, woutH, RES * RES);
    qkv_gemm_kernel<<<dim3(TOK / 64, QKVD / 64), 128, 0, stream>>>(xlnH, wqkvH, qkvH);
    vtrans_kernel<<<(BSZ * NH * HD * SEQ) / 256, 256, 0, stream>>>(qkvH, vtH);
    flash_kernel<<<dim3(SEQ / 128, NH, BSZ), 256, 0, stream>>>(qkvH, vtH, attnH);
    outproj_kernel<<<dim3(TOK / 64, RES / 64), 128, 0, stream>>>(attnH, woutH, x, out);
}
